// HeteHG_VAE_87686052315282
// MI455X (gfx1250) — matrix operand order repaired and checked by reading, not compiled, not run on hardware
//
#include <hip/hip_runtime.h>
#include <hip/hip_bf16.h>

typedef __attribute__((ext_vector_type(16))) _Float16 v16h;
typedef __attribute__((ext_vector_type(8)))  _Float16 v8h;
typedef __attribute__((ext_vector_type(8)))  float    v8f;

#define E_EDGES 4096
#define H1 512
#define H2 128
#define NNZ 131072
#define MTOT 13312   // 8192 + 4096 + 1024

// ---------------------------------------------------------------------------
// WMMA operand loaders (f16 16x16x32, wave32; contiguous k order, same for A and B)
// A 16x32: lanes 0-15 hold K[0..15], lanes 16-31 hold K[16..31]
// B 32x16 (col-major == row of Bt): lane n holds contiguous K chunk of 16
// ---------------------------------------------------------------------------
__device__ __forceinline__ v16h wmma_load_A(const _Float16* __restrict__ base,
                                            size_t row, int ld, int kk, int lane) {
    const _Float16* p = base + row * (size_t)ld + kk + ((lane >> 4) << 4);
    v8h lo = *(const v8h*)(p);
    v8h hi = *(const v8h*)(p + 8);
    v16h r;
#pragma unroll
    for (int i = 0; i < 8; ++i) { r[i] = lo[i]; r[i + 8] = hi[i]; }
    return r;
}
__device__ __forceinline__ v16h wmma_load_B(const _Float16* __restrict__ base,
                                            size_t row, int ld, int kk, int lane) {
    const _Float16* p = base + row * (size_t)ld + kk + ((lane >> 4) << 4);
    v8h lo = *(const v8h*)(p);
    v8h hi = *(const v8h*)(p + 8);
    v16h r;
#pragma unroll
    for (int i = 0; i < 8; ++i) { r[i] = lo[i]; r[i + 8] = hi[i]; }
    return r;
}

// ---------------------------------------------------------------------------
// Utility kernels
// ---------------------------------------------------------------------------
__global__ void zero_f32(float* __restrict__ p, long long n) {
    long long i = (long long)blockIdx.x * blockDim.x + threadIdx.x;
    long long stride = (long long)gridDim.x * blockDim.x;
    for (; i < n; i += stride) p[i] = 0.0f;
}

// W [K,N] f32 row-major  ->  Wt [N,K] f16 (K contiguous per row)
__global__ void transpose_cvt(const float* __restrict__ W, _Float16* __restrict__ Wt,
                              int K, int N) {
    long long i = (long long)blockIdx.x * blockDim.x + threadIdx.x;
    if (i >= (long long)K * N) return;
    int n = (int)(i % N);
    int k = (int)(i / N);
    Wt[(size_t)n * K + k] = (_Float16)W[i];
}

// COO scatter: h_pre[row] += val * Wn[col];  e_pre[col] += val * We[row]
// one thread per (nnz, h1) element; accumulators live in L2 (192MB)
__global__ __launch_bounds__(256) void scatter_inc(
    const int* __restrict__ rows, const int* __restrict__ cols,
    const float* __restrict__ vals,
    const float* __restrict__ Wn, const float* __restrict__ We,
    float* __restrict__ h_pre, float* __restrict__ e_pre) {
    size_t tid = (size_t)blockIdx.x * blockDim.x + threadIdx.x;
    int n = (int)(tid >> 9);
    int j = (int)(tid & 511);
    float v = vals[n];
    int r = rows[n], c = cols[n];
    atomicAdd(&h_pre[(size_t)r * H1 + j], v * Wn[(size_t)c * H1 + j]);
    atomicAdd(&e_pre[(size_t)c * H1 + j], v * We[(size_t)r * H1 + j]);
}

__global__ void tanh_bias_node(const float* __restrict__ in, const float* __restrict__ bias,
                               _Float16* __restrict__ out16, long long total) {
    long long i = (long long)blockIdx.x * blockDim.x + threadIdx.x;
    if (i >= total) return;
    int j = (int)(i & 511);
    out16[i] = (_Float16)tanhf(in[i] + bias[j]);
}

__global__ void tanh_bias_edge(const float* __restrict__ in, const float* __restrict__ bias,
                               float* __restrict__ out32, _Float16* __restrict__ out16,
                               long long total) {
    long long i = (long long)blockIdx.x * blockDim.x + threadIdx.x;
    if (i >= total) return;
    int j = (int)(i & 511);
    float v = tanhf(in[i] + bias[j]);
    out32[i] = v;
    out16[i] = (_Float16)v;
}

// ---------------------------------------------------------------------------
// Attention GEMM: C[M,N] = tanh(A[M,K]f16 @ Bt[N,K]f16^T + bias[N]) in f32
// one 16x16 tile per wave, 8 waves/block
// ---------------------------------------------------------------------------
__global__ __launch_bounds__(256) void gemm_bias_tanh(
    const _Float16* __restrict__ A, const _Float16* __restrict__ Bt,
    const float* __restrict__ bias, float* __restrict__ C,
    int M, int N, int K, int total_tiles) {
    int lane = threadIdx.x & 31;
    int wave = threadIdx.x >> 5;
    int t = blockIdx.x * 8 + wave;
    if (t >= total_tiles) return;
    int tiles_n = N >> 4;
    int tm = t / tiles_n;
    int tn = t - tm * tiles_n;
    size_t rowA = (size_t)(tm * 16 + (lane & 15));
    size_t rowB = (size_t)(tn * 16 + (lane & 15));
    v8f acc = {};
    for (int kk = 0; kk < K; kk += 32) {
        v16h a = wmma_load_A(A, rowA, K, kk, lane);
        v16h b = wmma_load_B(Bt, rowB, K, kk, lane);
        acc = __builtin_amdgcn_wmma_f32_16x16x32_f16(false, a, false, b, (short)0, acc,
                                                     false, false);
    }
    int col = tn * 16 + (lane & 15);
    int rbase = tm * 16 + ((lane >> 4) << 3);
    float bc = bias[col];
#pragma unroll
    for (int r = 0; r < 8; ++r)
        C[(size_t)(rbase + r) * N + col] = tanhf(acc[r] + bc);
}

// ---------------------------------------------------------------------------
// VAE head: zm = A@WmT^T + bm ; zs = A@WsT^T + bs ; z = zm + eps*exp(zs)
// outputs zm16, z16 as f16 [M,128].  K = 512, N = 128.
// ---------------------------------------------------------------------------
__global__ __launch_bounds__(256) void gemm_vae_head(
    const _Float16* __restrict__ A,
    const _Float16* __restrict__ BtM, const _Float16* __restrict__ BtS,
    const float* __restrict__ biasM, const float* __restrict__ biasS,
    const float* __restrict__ eps,
    _Float16* __restrict__ zm16, _Float16* __restrict__ z16, int M) {
    const int K = H1, N = H2;
    int lane = threadIdx.x & 31;
    int wave = threadIdx.x >> 5;
    int t = blockIdx.x * 8 + wave;          // tiles = (M/16)*8, exact
    int tm = t >> 3;
    int tn = t & 7;
    if (tm * 16 >= M) return;
    size_t rowA = (size_t)(tm * 16 + (lane & 15));
    size_t rowB = (size_t)(tn * 16 + (lane & 15));
    v8f accM = {}, accS = {};
    for (int kk = 0; kk < K; kk += 32) {
        v16h a  = wmma_load_A(A, rowA, K, kk, lane);
        v16h bm = wmma_load_B(BtM, rowB, K, kk, lane);
        v16h bs = wmma_load_B(BtS, rowB, K, kk, lane);
        accM = __builtin_amdgcn_wmma_f32_16x16x32_f16(false, a, false, bm, (short)0, accM,
                                                      false, false);
        accS = __builtin_amdgcn_wmma_f32_16x16x32_f16(false, a, false, bs, (short)0, accS,
                                                      false, false);
    }
    int col = tn * 16 + (lane & 15);
    int rbase = tm * 16 + ((lane >> 4) << 3);
    float bM = biasM[col], bS = biasS[col];
#pragma unroll
    for (int r = 0; r < 8; ++r) {
        size_t o = (size_t)(rbase + r) * N + col;
        float zm = accM[r] + bM;
        float zs = accS[r] + bS;
        float z  = zm + eps[o] * expf(zs);
        zm16[o] = (_Float16)zm;
        z16[o]  = (_Float16)z;
    }
}

// ---------------------------------------------------------------------------
// Attention score reduce + softmax mix
// ---------------------------------------------------------------------------
__global__ void score_reduce(const float* __restrict__ u, const float* __restrict__ uatt,
                             float* __restrict__ scores, int rows) {
    int i = blockIdx.x * blockDim.x + threadIdx.x;
    if (i >= rows) return;
    const float* up = u + (size_t)i * 256;
    float s = 0.f;
#pragma unroll 8
    for (int d = 0; d < 256; ++d) s += up[d] * uatt[d];
    scores[i] = s;
}

__global__ void attn_mix(const float* __restrict__ ea, const float* __restrict__ ek,
                         const float* __restrict__ ev, const float* __restrict__ scores,
                         _Float16* __restrict__ ee16) {
    long long i = (long long)blockIdx.x * blockDim.x + threadIdx.x; // E*512
    int e = (int)(i >> 9);
    float sa = scores[e], sk = scores[E_EDGES + e], sv = scores[2 * E_EDGES + e];
    float m = fmaxf(sa, fmaxf(sk, sv));
    float wa = expf(sa - m), wk = expf(sk - m), wv = expf(sv - m);
    float inv = 1.0f / (wa + wk + wv);
    float val = (wa * ea[i] + wk * ek[i] + wv * ev[i]) * inv;
    ee16[i] = (_Float16)val;
}

// ---------------------------------------------------------------------------
// Decoder: out[m, n] = Z[m,:] . ZE[n,:]   (K = 128, NCOL = 4096)
// each wave: 16 rows x 64 cols, A registers reused over 4 B tiles (16 WMMAs)
// ---------------------------------------------------------------------------
__global__ __launch_bounds__(256) void decoder_gemm(
    const _Float16* __restrict__ Z, const _Float16* __restrict__ ZE,
    float* __restrict__ out, unsigned long long out_base) {
    const int K = H2;          // 128
    const int NCOL = E_EDGES;  // 4096
    int lane = threadIdx.x & 31;
    int wave = threadIdx.x >> 5;
    int t = blockIdx.x * 8 + wave;     // (13312/16) * (4096/64) tiles, exact
    const int ngroups = NCOL >> 6;     // 64
    int tm = t / ngroups;
    int tg = t - tm * ngroups;
    int m0 = tm << 4;
    int n0 = tg << 6;
    size_t rowA = (size_t)(m0 + (lane & 15));
    v16h a0 = wmma_load_A(Z, rowA, K, 0,  lane);
    v16h a1 = wmma_load_A(Z, rowA, K, 32, lane);
    v16h a2 = wmma_load_A(Z, rowA, K, 64, lane);
    v16h a3 = wmma_load_A(Z, rowA, K, 96, lane);
    v8f acc[4];
#pragma unroll
    for (int jj = 0; jj < 4; ++jj) {
        size_t rowB = (size_t)(n0 + (jj << 4) + (lane & 15));
        v16h b0 = wmma_load_B(ZE, rowB, K, 0,  lane);
        v16h b1 = wmma_load_B(ZE, rowB, K, 32, lane);
        v16h b2 = wmma_load_B(ZE, rowB, K, 64, lane);
        v16h b3 = wmma_load_B(ZE, rowB, K, 96, lane);
        v8f c = {};
        c = __builtin_amdgcn_wmma_f32_16x16x32_f16(false, a0, false, b0, (short)0, c, false, false);
        c = __builtin_amdgcn_wmma_f32_16x16x32_f16(false, a1, false, b1, (short)0, c, false, false);
        c = __builtin_amdgcn_wmma_f32_16x16x32_f16(false, a2, false, b2, (short)0, c, false, false);
        c = __builtin_amdgcn_wmma_f32_16x16x32_f16(false, a3, false, b3, (short)0, c, false, false);
        acc[jj] = c;
    }
    int rbase = m0 + ((lane >> 4) << 3);
#pragma unroll
    for (int jj = 0; jj < 4; ++jj) {
        int col = n0 + (jj << 4) + (lane & 15);
#pragma unroll
        for (int r = 0; r < 8; ++r)
            out[out_base + (size_t)(rbase + r) * NCOL + col] = acc[jj][r];
    }
}

// ---------------------------------------------------------------------------
// Host orchestration
// ---------------------------------------------------------------------------
extern "C" void kernel_launch(void* const* d_in, const int* in_sizes, int n_in,
                              void* d_out, int out_size, void* d_ws, size_t ws_size,
                              hipStream_t stream) {
    (void)in_sizes; (void)n_in; (void)out_size; (void)ws_size;
    const int Ns[3]     = {8192, 4096, 1024};
    const int segOff[3] = {0, 8192, 12288};   // row offsets into stacked node arrays

    // ---- workspace carve-up (256B aligned) ----
    char* ws = (char*)d_ws;
    size_t off = 0;
    auto take = [&](size_t bytes) -> size_t {
        size_t o = off;
        off += (bytes + 255) & ~(size_t)255;
        return o;
    };
    float*    h_pre = (float*)   (ws + take((size_t)MTOT * H1 * 4));
    float*    e_pre = (float*)   (ws + take((size_t)3 * E_EDGES * H1 * 4));
    _Float16* h16   = (_Float16*)(ws + take((size_t)MTOT * H1 * 2));
    _Float16* e16   = (_Float16*)(ws + take((size_t)3 * E_EDGES * H1 * 2));
    _Float16* WzmT  = (_Float16*)(ws + take((size_t)3 * H2 * H1 * 2));
    _Float16* WzsT  = (_Float16*)(ws + take((size_t)3 * H2 * H1 * 2));
    _Float16* WattT = (_Float16*)(ws + take((size_t)256 * H1 * 2));
    _Float16* WemT  = (_Float16*)(ws + take((size_t)H2 * H1 * 2));
    _Float16* WesT  = (_Float16*)(ws + take((size_t)H2 * H1 * 2));
    _Float16* ee16  = (_Float16*)(ws + take((size_t)E_EDGES * H1 * 2));
    _Float16* z16   = (_Float16*)(ws + take((size_t)MTOT * H2 * 2));
    _Float16* zm16  = (_Float16*)(ws + take((size_t)MTOT * H2 * 2));
    _Float16* ze16  = (_Float16*)(ws + take((size_t)E_EDGES * H2 * 2));
    _Float16* zem16 = (_Float16*)(ws + take((size_t)E_EDGES * H2 * 2));
    // u_pre / scores alias the h_pre region (h_pre dead after node tanh)
    float* u_pre  = h_pre;
    float* scores = (float*)((char*)h_pre + (size_t)3 * E_EDGES * 256 * 4);

    float* out = (float*)d_out;

    // ---- input pointers (setup_inputs dict order) ----
    const int*   rows[3]; const int* cols[3]; const float* vals[3];
    const float *Wn[3], *bn[3], *Wzm[3], *bzm[3], *Wzs[3], *bzs[3], *We[3], *be[3], *eps[3];
    for (int t = 0; t < 3; ++t) {
        int b = t * 12;
        rows[t] = (const int*)  d_in[b + 0];
        cols[t] = (const int*)  d_in[b + 1];
        vals[t] = (const float*)d_in[b + 2];
        Wn[t]   = (const float*)d_in[b + 3];
        bn[t]   = (const float*)d_in[b + 4];
        Wzm[t]  = (const float*)d_in[b + 5];
        bzm[t]  = (const float*)d_in[b + 6];
        Wzs[t]  = (const float*)d_in[b + 7];
        bzs[t]  = (const float*)d_in[b + 8];
        We[t]   = (const float*)d_in[b + 9];
        be[t]   = (const float*)d_in[b + 10];
        eps[t]  = (const float*)d_in[b + 11];
    }
    const float* Watt  = (const float*)d_in[36];
    const float* batt  = (const float*)d_in[37];
    const float* uatt  = (const float*)d_in[38];
    const float* Wem   = (const float*)d_in[39];
    const float* bem   = (const float*)d_in[40];
    const float* Wes   = (const float*)d_in[41];
    const float* bes   = (const float*)d_in[42];
    const float* eps_e = (const float*)d_in[43];

    // ---- 1) zero accumulators ----
    zero_f32<<<2048, 256, 0, stream>>>(h_pre, (long long)MTOT * H1);
    zero_f32<<<2048, 256, 0, stream>>>(e_pre, (long long)3 * E_EDGES * H1);

    // ---- 2) COO segment-sum scatter (atomics, accumulators L2-resident) ----
    for (int t = 0; t < 3; ++t) {
        scatter_inc<<<(NNZ * H1) / 256, 256, 0, stream>>>(
            rows[t], cols[t], vals[t], Wn[t], We[t],
            h_pre + (size_t)segOff[t] * H1,
            e_pre + (size_t)t * E_EDGES * H1);
    }

    // ---- 3) weight transpose + f32->f16 (B operands want [N,K] K-contiguous) ----
    for (int t = 0; t < 3; ++t) {
        transpose_cvt<<<(H1 * H2 + 255) / 256, 256, 0, stream>>>(Wzm[t], WzmT + (size_t)t * H2 * H1, H1, H2);
        transpose_cvt<<<(H1 * H2 + 255) / 256, 256, 0, stream>>>(Wzs[t], WzsT + (size_t)t * H2 * H1, H1, H2);
    }
    transpose_cvt<<<(H1 * 256 + 255) / 256, 256, 0, stream>>>(Watt, WattT, H1, 256);
    transpose_cvt<<<(H1 * H2 + 255) / 256, 256, 0, stream>>>(Wem, WemT, H1, H2);
    transpose_cvt<<<(H1 * H2 + 255) / 256, 256, 0, stream>>>(Wes, WesT, H1, H2);

    // ---- 4) tanh + bias; f16 copies for WMMA ----
    for (int t = 0; t < 3; ++t) {
        long long tot = (long long)Ns[t] * H1;
        tanh_bias_node<<<(unsigned)((tot + 255) / 256), 256, 0, stream>>>(
            h_pre + (size_t)segOff[t] * H1, bn[t], h16 + (size_t)segOff[t] * H1, tot);
        long long tote = (long long)E_EDGES * H1;
        tanh_bias_edge<<<(unsigned)((tote + 255) / 256), 256, 0, stream>>>(
            e_pre + (size_t)t * E_EDGES * H1, be[t],
            e_pre + (size_t)t * E_EDGES * H1, e16 + (size_t)t * E_EDGES * H1, tote);
    }

    // ---- 5) node VAE heads: zm, z (WMMA, K=512) ----
    for (int t = 0; t < 3; ++t) {
        gemm_vae_head<<<Ns[t] / 16, 256, 0, stream>>>(
            h16 + (size_t)segOff[t] * H1,
            WzmT + (size_t)t * H2 * H1, WzsT + (size_t)t * H2 * H1,
            bzm[t], bzs[t], eps[t],
            zm16 + (size_t)segOff[t] * H2, z16 + (size_t)segOff[t] * H2, Ns[t]);
    }

    // ---- 6) attention: u = tanh(e16 @ WattT^T + batt), [12288, 256] ----
    {
        int M = 3 * E_EDGES, N = 256, K = H1;
        int tiles = (M / 16) * (N / 16);           // 12288
        gemm_bias_tanh<<<tiles / 8, 256, 0, stream>>>(e16, WattT, batt, u_pre, M, N, K, tiles);
    }

    // ---- 7) scores + softmax mix -> edge_embed f16 ----
    score_reduce<<<(3 * E_EDGES + 255) / 256, 256, 0, stream>>>(u_pre, uatt, scores, 3 * E_EDGES);
    attn_mix<<<(E_EDGES * H1) / 256, 256, 0, stream>>>(
        e_pre, e_pre + (size_t)E_EDGES * H1, e_pre + (size_t)2 * E_EDGES * H1, scores, ee16);

    // ---- 8) edge VAE head: zem, ze ----
    gemm_vae_head<<<E_EDGES / 16, 256, 0, stream>>>(
        ee16, WemT, WesT, bem, bes, eps_e, zem16, ze16, E_EDGES);

    // ---- 9) decoders: z @ ze^T then zm @ zem^T (WMMA f16, f32 accum) ----
    {
        int blocks = (MTOT / 16) * (E_EDGES / 64) / 8;   // 6656
        decoder_gemm<<<blocks, 256, 0, stream>>>(z16, ze16, out, 0ULL);
        decoder_gemm<<<blocks, 256, 0, stream>>>(zm16, zem16, out,
                                                 (unsigned long long)MTOT * E_EDGES);
    }
}
